// GroupedQueryAttention_45019847197304
// MI455X (gfx1250) — matrix operand order repaired and checked by reading, not compiled, not run on hardware
//
#include <hip/hip_runtime.h>
#include <hip/hip_bf16.h>

// ---------------------------------------------------------------------------
// Types / WMMA helpers (CDNA5 gfx1250, wave32)
// ---------------------------------------------------------------------------
typedef __attribute__((ext_vector_type(16))) _Float16 v16h;
typedef __attribute__((ext_vector_type(8)))  float    v8f;
typedef unsigned int u32x4 __attribute__((ext_vector_type(4)));
typedef int          i32x8 __attribute__((ext_vector_type(8)));
typedef int          i32x4 __attribute__((ext_vector_type(4)));

#if defined(__gfx1250__) && __has_builtin(__builtin_amdgcn_tensor_load_to_lds) && \
    __has_builtin(__builtin_amdgcn_s_wait_tensorcnt)
#define HAVE_TDM 1
#else
#define HAVE_TDM 0
#endif

union FragU { v16h h; unsigned u[8]; };

// A-matrix 16x32 f16 fragment from LDS tile laid out [row][K] (ld in f16 units).
// ISA 7.12.2: lane<16 -> M=lane, K=0..7,16..23 ; lane>=16 -> K=8..15,24..31.
__device__ __forceinline__ v16h ld_fragA(const _Float16* base, int row, int ld, int lane) {
  const unsigned* p = (const unsigned*)(base + (size_t)row * ld);
  int d0 = (lane >> 4) * 4;
  FragU f;
#pragma unroll
  for (int i = 0; i < 4; ++i) { f.u[i] = p[d0 + i]; f.u[4 + i] = p[d0 + 8 + i]; }
  return f.h;
}

// B-matrix 32x16 f16 fragment from LDS tile laid out [N][K]:
// lane<16 -> N=lane, K=0..7,16..23 ; lane>=16 -> K=8..15,24..31.
__device__ __forceinline__ v16h ld_fragB(const _Float16* base, int row, int ld, int lane) {
  const unsigned* p = (const unsigned*)(base + (size_t)row * ld);
  int d0 = (lane >> 4) * 4;
  FragU f;
#pragma unroll
  for (int i = 0; i < 4; ++i) { f.u[i] = p[d0 + i]; f.u[4 + i] = p[d0 + 8 + i]; }
  return f.h;
}

__device__ __forceinline__ v8f wmma_f16(v16h a, v16h b, v8f c) {
  return __builtin_amdgcn_wmma_f32_16x16x32_f16(false, a, false, b, (short)0, c, false, false);
}

// ---------------------------------------------------------------------------
// Tensor Data Mover: 2D tile global->LDS with hardware LDS padding.
// D# per ISA ch8: group0 = {flags, lds_addr, global_addr, type=2};
// group1 = {data_size=2B, pad fields, tensor dims, tile dims, dim0 stride}.
//   pad_interval code: interval dwords = 2 << code ; pad_amount code: dwords = code+1
// clang-23 builtin form: (u32x4, i32x8, i32x4, i32x4, i32x8, i32 cpol)
// ---------------------------------------------------------------------------
#if HAVE_TDM
__device__ __forceinline__ void tdm_load_2d(const void* lds_dst, const _Float16* gsrc,
                                            unsigned tile_k, unsigned tile_rows,
                                            unsigned row_stride_elems,
                                            unsigned pad_int_code, unsigned pad_amt_code) {
  unsigned long long ga = (unsigned long long)(const void*)gsrc;
  unsigned lds = (unsigned)(size_t)lds_dst;   // flat LDS addr low bits = LDS offset
  u32x4 g0;
  g0[0] = 1u;                                           // count=1, user descriptor
  g0[1] = lds;                                          // lds_addr [63:32]
  g0[2] = (unsigned)(ga & 0xffffffffu);                 // global_addr low
  g0[3] = (unsigned)((ga >> 32) & 0x1ffffffu) | (2u << 30); // global_addr hi + type=2
  i32x8 g1;
  unsigned td0 = row_stride_elems;                      // tensor_dim0
  unsigned td1 = tile_rows;                             // tensor_dim1
  g1[0] = (int)((1u << 16)                              // data_size = 2 bytes
              | (1u << 20)                              // pad_enable
              | (pad_int_code << 22) | (pad_amt_code << 25));
  g1[1] = (int)((td0 & 0xffffu) << 16);                 // td0[15:0] at bits 63:48
  g1[2] = (int)(((td0 >> 16) & 0xffffu) | ((td1 & 0xffffu) << 16));
  g1[3] = (int)(((td1 >> 16) & 0xffffu) | ((tile_k & 0xffffu) << 16)); // tile_dim0
  g1[4] = (int)(tile_rows & 0xffffu);                   // tile_dim1 (tile_dim2=0)
  g1[5] = (int)row_stride_elems;                        // tensor_dim0_stride low32
  g1[6] = 0;
  g1[7] = 0;
  i32x4 z4 = {0, 0, 0, 0};
  i32x8 z8 = {0, 0, 0, 0, 0, 0, 0, 0};
  __builtin_amdgcn_tensor_load_to_lds(g0, g1, z4, z4, z8, 0);
}
#endif

// ---------------------------------------------------------------------------
// f32 -> f16 cast (grid-stride)
// ---------------------------------------------------------------------------
__global__ __launch_bounds__(256) void cast_f16_kernel(const float* __restrict__ x,
                                                       _Float16* __restrict__ y, int n) {
  for (int i = blockIdx.x * blockDim.x + threadIdx.x; i < n; i += gridDim.x * blockDim.x)
    y[i] = (_Float16)x[i];
}

// f32 RxC row-major -> f16 CxR row-major (weights pre-transposed to [N][K])
__global__ __launch_bounds__(256) void cast_transpose_kernel(const float* __restrict__ x,
                                                             _Float16* __restrict__ y,
                                                             int R, int C) {
  __shared__ float tile[32][33];
  int c0 = blockIdx.x * 32, r0 = blockIdx.y * 32;
  int tx = threadIdx.x & 31, ty = threadIdx.x >> 5;
#pragma unroll
  for (int i = ty; i < 32; i += 8)
    tile[i][tx] = x[(size_t)(r0 + i) * C + c0 + tx];
  __syncthreads();
#pragma unroll
  for (int i = ty; i < 32; i += 8)
    y[(size_t)(c0 + i) * R + r0 + tx] = (_Float16)tile[tx][i];
}

// ---------------------------------------------------------------------------
// WMMA GEMM: C(f32, MxN) = A(f16, MxK rm) @ Bt(f16, NxK rm, pre-transposed) + bias
// Block: 256 threads = 8 waves. Tile M=128, N=64, K-step 64.
// Double-buffered TDM staging: issue next tile pair, s_wait_tensorcnt(2) so the
// current pair has landed (TDM completes in order per wave), barrier, compute.
// ---------------------------------------------------------------------------
__global__ __launch_bounds__(256) void gemm_f16f32(const _Float16* __restrict__ A,
                                                   const _Float16* __restrict__ Bt,
                                                   const float* __restrict__ bias,
                                                   float* __restrict__ C,
                                                   int M, int N, int K) {
  __shared__ _Float16 sA[2][128][72];  // 64-f16 rows + 8 f16 pad (32 dw + 4 dw)
  __shared__ _Float16 sB[2][64][72];
  const int t = threadIdx.x, lane = t & 31, w = t >> 5;
  const int m0 = blockIdx.y * 128, n0 = blockIdx.x * 64;
  const int col15 = lane & 15, hi = lane >> 4;

  v8f acc[4] = {};
#if HAVE_TDM
  if (w == 0) {
    tdm_load_2d(&sA[0][0][0], &A[(size_t)m0 * K], 64, 128, K, /*32dw*/4, /*4dw*/3);
    tdm_load_2d(&sB[0][0][0], &Bt[(size_t)n0 * K], 64, 64, K, 4, 3);
  }
#endif
  for (int k0 = 0; k0 < K; k0 += 64) {
    const int cur = (k0 >> 6) & 1;
#if HAVE_TDM
    if (w == 0) {
      if (k0 + 64 < K) {
        tdm_load_2d(&sA[cur ^ 1][0][0], &A[(size_t)m0 * K + k0 + 64], 64, 128, K, 4, 3);
        tdm_load_2d(&sB[cur ^ 1][0][0], &Bt[(size_t)n0 * K + k0 + 64], 64, 64, K, 4, 3);
        __builtin_amdgcn_s_wait_tensorcnt(2);   // current pair complete, next in flight
      } else {
        __builtin_amdgcn_s_wait_tensorcnt(0);
      }
    }
#else
#pragma unroll
    for (int j = 0; j < 4; ++j) {    // A tile: 128x64 f16, 4x16B per thread
      int i = t + 256 * j;
      int r = i >> 3, cs = (i & 7) * 8;
      *(float4*)&sA[cur][r][cs] = *(const float4*)&A[(size_t)(m0 + r) * K + k0 + cs];
    }
#pragma unroll
    for (int j = 0; j < 2; ++j) {    // B tile: 64x64 f16
      int i = t + 256 * j;
      int r = i >> 3, cs = (i & 7) * 8;
      *(float4*)&sB[cur][r][cs] = *(const float4*)&Bt[(size_t)(n0 + r) * K + k0 + cs];
    }
#endif
    __syncthreads();
    v16h a0 = ld_fragA(&sA[cur][0][0], w * 16 + col15, 72, lane);       // K 0..31
    v16h a1 = ld_fragA(&sA[cur][0][0] + 32, w * 16 + col15, 72, lane);  // K 32..63
#pragma unroll
    for (int nb = 0; nb < 4; ++nb) {
      v16h b0 = ld_fragB(&sB[cur][0][0], nb * 16 + col15, 72, lane);
      v16h b1 = ld_fragB(&sB[cur][0][0] + 32, nb * 16 + col15, 72, lane);
      acc[nb] = wmma_f16(a0, b0, acc[nb]);
      acc[nb] = wmma_f16(a1, b1, acc[nb]);
    }
    __syncthreads();
  }
#pragma unroll
  for (int nb = 0; nb < 4; ++nb) {
    int col = n0 + nb * 16 + col15;
    float bv = bias[col];
#pragma unroll
    for (int r = 0; r < 8; ++r) {
      int row = m0 + w * 16 + r + 8 * hi;
      C[(size_t)row * N + col] = acc[nb][r] + bv;
    }
  }
}

// ---------------------------------------------------------------------------
// Fused RoPE (full-row halves, as reference) + per-head(64) RMSNorm + repack
// to (B, NH, S, 64) f16. oscale folds 1/sqrt(d) into q.
// ---------------------------------------------------------------------------
__global__ __launch_bounds__(256) void rope_rms_kernel(const float* __restrict__ X,
                                                       _Float16* __restrict__ out,
                                                       int Dtot, int half, int NH, int S,
                                                       float oscale) {
  __shared__ float ps[256];
  __shared__ float hsum[16];
  const int bs = blockIdx.x;
  const int b = bs / S, s = bs - b * S;
  const int t = threadIdx.x;
  const int epr = Dtot >> 8;     // 4 (q) or 1 (k)
  const int e0 = t * epr;
  const float* row = X + (size_t)bs * Dtot;

  float vals[4];
  float ss = 0.f;
  for (int j = 0; j < epr; ++j) {
    int e = e0 + j;
    int jj = (e < half) ? e : e - half;
    float freq = __expf((float)jj * (-9.210340371976184f / (float)half));
    float ang = (float)s * freq;
    float sn, c;
    __sincosf(ang, &sn, &c);
    float v = (e < half) ? (row[e] * c - row[e + half] * sn)
                         : (row[e] * c + row[e - half] * sn);
    vals[j] = v;
    ss += v * v;
  }
  ps[t] = ss;
  __syncthreads();
  const int tph = 256 / NH;
  if (t < NH) {
    float acc = 0.f;
    for (int i = 0; i < tph; ++i) acc += ps[t * tph + i];
    hsum[t] = acc;
  }
  __syncthreads();
  const int h = e0 >> 6;
  const float nrm = rsqrtf(hsum[h] * (1.0f / 64.0f) + 1.1920929e-7f) * oscale;
  for (int j = 0; j < epr; ++j) {
    int e = e0 + j;
    out[(((size_t)b * NH + h) * S + s) * 64 + (e & 63)] = (_Float16)(vals[j] * nrm);
  }
}

// (B,S,256) f32 -> (B,4,S,64) f16
__global__ __launch_bounds__(256) void reorder_v_kernel(const float* __restrict__ v,
                                                        _Float16* __restrict__ out,
                                                        int Bn, int S) {
  int n = Bn * S * 256;
  for (int i = blockIdx.x * blockDim.x + threadIdx.x; i < n; i += gridDim.x * blockDim.x) {
    int b = i / (S * 256);
    int rem = i - b * S * 256;
    int s = rem >> 8;
    int e = rem & 255;
    int h = e >> 6, d = e & 63;
    out[(((size_t)b * 4 + h) * S + s) * 64 + d] = (_Float16)v[i];
  }
}

// ---------------------------------------------------------------------------
// Flash attention, one wave (32 threads) per (b, head, 16 q-rows).
// Q pre-scaled by 1/8. Keys processed in chunks of 32 with online softmax.
// K tiles double-buffered through the TDM; V transposed manually ([d][key]).
// ---------------------------------------------------------------------------
__global__ __launch_bounds__(32) void attn_kernel(const _Float16* __restrict__ Q,   // (B,16,S,64)
                                                  const _Float16* __restrict__ Kh,  // (B,4,S,64)
                                                  const _Float16* __restrict__ Vh,  // (B,4,S,64)
                                                  const float* __restrict__ mask,   // (B,S,S)
                                                  _Float16* __restrict__ O,         // (B,S,1024)
                                                  int S) {
  __shared__ _Float16 sQ[16][72];      // 64-f16 rows + 8 f16 pad
  __shared__ _Float16 sK[2][32][72];   // double-buffered K tiles
  __shared__ _Float16 sV[64][40];      // transposed: [d][key]
  __shared__ _Float16 sP[16][40];

  const int lane = threadIdx.x;
  const int col15 = lane & 15, hi = lane >> 4;
  const int bh = blockIdx.y;
  const int b = bh >> 4, h = bh & 15, kvh = h >> 2;
  const int qt = blockIdx.x;

  const _Float16* qg = Q + ((size_t)(b * 16 + h) * S + qt * 16) * 64;
  const _Float16* kbase = Kh + ((size_t)(b * 4 + kvh) * S) * 64;
  const _Float16* vbase = Vh + ((size_t)(b * 4 + kvh) * S) * 64;
  const float* mbase = mask + ((size_t)b * S + qt * 16) * S;

#if HAVE_TDM
  tdm_load_2d(&sQ[0][0], qg, 64, 16, 64, /*32dw*/4, /*4dw*/3);
  tdm_load_2d(&sK[0][0][0], kbase, 64, 32, 64, 4, 3);
  __builtin_amdgcn_s_wait_tensorcnt(0);
#else
#pragma unroll
  for (int j = 0; j < 4; ++j) {
    int o = lane + 32 * j;
    int r = o >> 3, cs = (o & 7) * 8;
    *(float4*)&sQ[r][cs] = *(const float4*)&qg[(size_t)r * 64 + cs];
  }
#endif
  __syncthreads();
  const v16h aQ0 = ld_fragA(&sQ[0][0], col15, 72, lane);       // d 0..31
  const v16h aQ1 = ld_fragA(&sQ[0][0] + 32, col15, 72, lane);  // d 32..63

  float mrow[8], lrow[8];
  v8f accO[4] = {};
#pragma unroll
  for (int r = 0; r < 8; ++r) { mrow[r] = -1e30f; lrow[r] = 0.f; }

  for (int kc = 0; kc < S; kc += 32) {
    const int cur = (kc >> 5) & 1;
    // V chunk transposed into [d][key]
#pragma unroll
    for (int j = 0; j < 8; ++j) {
      int o = lane + 32 * j;
      int r = o >> 3, cs = (o & 7) * 8;
#if !HAVE_TDM
      *(float4*)&sK[cur][r][cs] = *(const float4*)&kbase[((size_t)(kc + r)) * 64 + cs];
#endif
      _Float16 tmp[8];
      *(float4*)tmp = *(const float4*)&vbase[((size_t)(kc + r)) * 64 + cs];
#pragma unroll
      for (int e = 0; e < 8; ++e) sV[cs + e][r] = tmp[e];
    }
#if HAVE_TDM
    if (kc + 32 < S) {
      tdm_load_2d(&sK[cur ^ 1][0][0], &kbase[(size_t)(kc + 32) * 64], 64, 32, 64, 4, 3);
      __builtin_amdgcn_s_wait_tensorcnt(1);   // current K tile complete, next in flight
    } else {
      __builtin_amdgcn_s_wait_tensorcnt(0);
    }
#endif
    __syncthreads();

    // scores: two 16x16 tiles (keys kc+0..15, kc+16..31), K-dim = 64 (two WMMAs)
    v8f sc[2];
#pragma unroll
    for (int nb = 0; nb < 2; ++nb) {
      v16h bk0 = ld_fragB(&sK[cur][0][0], nb * 16 + col15, 72, lane);
      v16h bk1 = ld_fragB(&sK[cur][0][0] + 32, nb * 16 + col15, 72, lane);
      v8f z = {};
      z = wmma_f16(aQ0, bk0, z);
      z = wmma_f16(aQ1, bk1, z);
      int key = kc + nb * 16 + col15;
      const float* mp = mbase + key;
#pragma unroll
      for (int r = 0; r < 8; ++r)
        z[r] += mp[(size_t)(r + 8 * hi) * S];
      sc[nb] = z;
    }

    // online softmax (row reductions across the 16 lanes of each half-wave)
    float scl[8];
#pragma unroll
    for (int r = 0; r < 8; ++r) {
      float mx = fmaxf(sc[0][r], sc[1][r]);
#pragma unroll
      for (int off = 8; off >= 1; off >>= 1) mx = fmaxf(mx, __shfl_xor(mx, off, 32));
      float mnew = fmaxf(mrow[r], mx);
      float sc_old = __expf(mrow[r] - mnew);
      float p0 = __expf(sc[0][r] - mnew);
      float p1 = __expf(sc[1][r] - mnew);
      sc[0][r] = p0; sc[1][r] = p1;
      float psum = p0 + p1;
#pragma unroll
      for (int off = 8; off >= 1; off >>= 1) psum += __shfl_xor(psum, off, 32);
      lrow[r] = lrow[r] * sc_old + psum;
      mrow[r] = mnew;
      scl[r] = sc_old;
    }

    // write P tile (16x32) to LDS to re-fragment as A
#pragma unroll
    for (int nb = 0; nb < 2; ++nb)
#pragma unroll
      for (int r = 0; r < 8; ++r)
        sP[r + 8 * hi][nb * 16 + col15] = (_Float16)sc[nb][r];
    __syncthreads();

    v16h aP = ld_fragA(&sP[0][0], col15, 40, lane);
#pragma unroll
    for (int nbd = 0; nbd < 4; ++nbd) {
      v16h bv = ld_fragB(&sV[0][0], nbd * 16 + col15, 40, lane);
      v8f c = accO[nbd];
#pragma unroll
      for (int r = 0; r < 8; ++r) c[r] *= scl[r];
      accO[nbd] = wmma_f16(aP, bv, c);
    }
    __syncthreads();
  }

#pragma unroll
  for (int nbd = 0; nbd < 4; ++nbd) {
#pragma unroll
    for (int r = 0; r < 8; ++r) {
      int row = qt * 16 + r + 8 * hi;
      int dcol = nbd * 16 + col15;
      O[((size_t)b * S + row) * 1024 + h * 64 + dcol] = (_Float16)(accO[nbd][r] / lrow[r]);
    }
  }
}

// ---------------------------------------------------------------------------
// Host launch
// ---------------------------------------------------------------------------
extern "C" void kernel_launch(void* const* d_in, const int* in_sizes, int n_in,
                              void* d_out, int out_size, void* d_ws, size_t ws_size,
                              hipStream_t stream) {
  (void)in_sizes; (void)n_in; (void)out_size; (void)ws_size;
  constexpr int Bn = 2, S = 2048, HID = 1024, NH = 16, NKV = 4, KVD = 256;
  const int M = Bn * S;

  const float* hidden = (const float*)d_in[0];
  const float* amask  = (const float*)d_in[1];
  const float* Wq = (const float*)d_in[2];
  const float* bq = (const float*)d_in[3];
  const float* Wk = (const float*)d_in[4];
  const float* bk = (const float*)d_in[5];
  const float* Wv = (const float*)d_in[6];
  const float* bv = (const float*)d_in[7];
  const float* Wo = (const float*)d_in[8];
  const float* bo = (const float*)d_in[9];

  char* ws = (char*)d_ws;
  size_t off = 0;
  auto take = [&](size_t bytes) -> char* {
    char* p = ws + off;
    off = (off + bytes + 255) & ~(size_t)255;
    return p;
  };
  _Float16* hs16  = (_Float16*)take((size_t)M * HID * 2);
  _Float16* wq16t = (_Float16*)take((size_t)HID * HID * 2);   // [N][K]
  _Float16* wk16t = (_Float16*)take((size_t)HID * KVD * 2);
  _Float16* wv16t = (_Float16*)take((size_t)HID * KVD * 2);
  _Float16* wo16t = (_Float16*)take((size_t)HID * HID * 2);
  float*    qf32  = (float*)take((size_t)M * HID * 4);
  float*    kf32  = (float*)take((size_t)M * KVD * 4);
  float*    vf32  = (float*)take((size_t)M * KVD * 4);
  _Float16* qh16  = (_Float16*)take((size_t)M * HID * 2);
  _Float16* kh16  = (_Float16*)take((size_t)M * KVD * 2);
  _Float16* vh16  = (_Float16*)take((size_t)M * KVD * 2);
  _Float16* at16  = (_Float16*)take((size_t)M * HID * 2);

  // 1) cast activations; cast+transpose weights to [N][K]
  cast_f16_kernel<<<2048, 256, 0, stream>>>(hidden, hs16, M * HID);
  cast_transpose_kernel<<<dim3(HID / 32, HID / 32), 256, 0, stream>>>(Wq, wq16t, HID, HID);
  cast_transpose_kernel<<<dim3(KVD / 32, HID / 32), 256, 0, stream>>>(Wk, wk16t, HID, KVD);
  cast_transpose_kernel<<<dim3(KVD / 32, HID / 32), 256, 0, stream>>>(Wv, wv16t, HID, KVD);
  cast_transpose_kernel<<<dim3(HID / 32, HID / 32), 256, 0, stream>>>(Wo, wo16t, HID, HID);

  // 2) QKV projections (WMMA, double-buffered TDM tiles)
  gemm_f16f32<<<dim3(HID / 64, M / 128), 256, 0, stream>>>(hs16, wq16t, bq, qf32, M, HID, HID);
  gemm_f16f32<<<dim3(KVD / 64, M / 128), 256, 0, stream>>>(hs16, wk16t, bk, kf32, M, KVD, HID);
  gemm_f16f32<<<dim3(KVD / 64, M / 128), 256, 0, stream>>>(hs16, wv16t, bv, vf32, M, KVD, HID);

  // 3) RoPE + RMSNorm + head-major repack (q folded with 1/sqrt(64))
  rope_rms_kernel<<<M, 256, 0, stream>>>(qf32, qh16, HID, HID / 2, NH, S, 0.125f);
  rope_rms_kernel<<<M, 256, 0, stream>>>(kf32, kh16, KVD, KVD / 2, NKV, S, 1.0f);
  reorder_v_kernel<<<1024, 256, 0, stream>>>(vf32, vh16, Bn, S);

  // 4) flash attention (WMMA, online softmax, double-buffered TDM K tiles)
  attn_kernel<<<dim3(S / 16, Bn * NH), 32, 0, stream>>>(qh16, kh16, vh16, amask, at16, S);

  // 5) output projection (WMMA) -> f32 d_out
  gemm_f16f32<<<dim3(HID / 64, M / 128), 256, 0, stream>>>(at16, wo16t, bo, (float*)d_out, M, HID, HID);
}
